// QuantizedLinear_70385924047666
// MI455X (gfx1250) — compile-verified, matrix operand order checked by reading, not run on hardware
//
#include <hip/hip_runtime.h>

// ---------------------------------------------------------------------------
// BitNet quantized linear  ->  f16 WMMA GEMM on gfx1250 (MI455X)
//   out[m,n] = scale[n] * sum_k x[m,k]*qw[n,k] + bias[n]
//   M = 8192, N = 11008, K = 4096
// Weights are small ints (exact in f16); x converted f32->f16 on the fly,
// accumulation in f32 via v_wmma_f32_16x16x32_f16, scale/bias epilogue in f32.
// Global loads for K-tile k+1 are software-pipelined into registers while the
// WMMAs for tile k execute (removes the serialized load/wait pattern).
// ---------------------------------------------------------------------------

typedef _Float16 half4  __attribute__((ext_vector_type(4)));
typedef _Float16 half8  __attribute__((ext_vector_type(8)));
typedef _Float16 half16 __attribute__((ext_vector_type(16)));
typedef float    f32x4  __attribute__((ext_vector_type(4)));
typedef float    f32x8  __attribute__((ext_vector_type(8)));
typedef int      i32x4  __attribute__((ext_vector_type(4)));

#define BLK_M      128   // block tile M  (2 waves)
#define BLK_N      256   // block tile N  (4 waves)
#define BLK_K      32    // K per WMMA step
#define LDS_STRIDE 40    // 32 halves + 8 pad -> 80B row stride (bank-conflict-minimal)

__global__ __launch_bounds__(256, 1)
void bitnet_wmma_f16_kernel(const float* __restrict__ x,
                            const int*   __restrict__ qw,
                            const float* __restrict__ scale,
                            const float* __restrict__ bias,
                            float*       __restrict__ out,
                            int M, int N, int K)
{
    __shared__ _Float16 lds_a[BLK_M * LDS_STRIDE];   // x tile   (f16)
    __shared__ _Float16 lds_b[BLK_N * LDS_STRIDE];   // qw tile  (f16, exact)

    const int tid  = threadIdx.x;
    const int lane = tid & 31;
    const int wave = tid >> 5;          // 0..7
    const int wm   = wave & 1;          // wave M coord (0..1)
    const int wn   = wave >> 1;         // wave N coord (0..3)

    const int m0 = blockIdx.y * BLK_M;
    const int n0 = blockIdx.x * BLK_N;

    // WMMA fragment addressing (16-bit A 16x32 / B 32x16, same k order used for A and B):
    // lane L -> row/col L%16, contiguous K halves starting at (L/16)*16.
    const int r  = lane & 15;
    const int kh = lane >> 4;           // 0 or 1 -> K offset 0 / 16

    // ---- fixed per-thread staging coordinates --------------------------------
    // A tile: 128 rows x 32 f32 -> 1024 float4; 4 per thread.
    // B tile: 256 rows x 32 i32 -> 2048 int4;   8 per thread.
    int a_row[4], a_c4[4], b_row[8], b_c4[8];
    const float* a_ptr[4];
    const int*   b_ptr[8];
    #pragma unroll
    for (int i = 0; i < 4; ++i) {
        int idx  = tid + 256 * i;
        a_row[i] = idx >> 3;
        a_c4[i]  = (idx & 7) << 2;
        a_ptr[i] = x + (size_t)(m0 + a_row[i]) * K + a_c4[i];
    }
    #pragma unroll
    for (int i = 0; i < 8; ++i) {
        int idx  = tid + 256 * i;
        b_row[i] = idx >> 3;
        b_c4[i]  = (idx & 7) << 2;
        b_ptr[i] = qw + (size_t)(n0 + b_row[i]) * K + b_c4[i];
    }

    f32x8 acc[4][4] = {};               // 64x64 per-wave C tile, f32

    // ---- prologue: issue K-tile 0 loads into registers ----------------------
    f32x4 areg[4];
    i32x4 breg[8];
    #pragma unroll
    for (int i = 0; i < 4; ++i) areg[i] = *(const f32x4*)(a_ptr[i]);
    #pragma unroll
    for (int i = 0; i < 8; ++i) breg[i] = *(const i32x4*)(b_ptr[i]);

    for (int k0 = 0; k0 < K; k0 += BLK_K) {
        // ---- convert staged registers -> LDS (f16) --------------------------
        #pragma unroll
        for (int i = 0; i < 4; ++i) {
            half4 h = { (_Float16)areg[i][0], (_Float16)areg[i][1],
                        (_Float16)areg[i][2], (_Float16)areg[i][3] };
            *(half4*)(&lds_a[a_row[i] * LDS_STRIDE + a_c4[i]]) = h;
        }
        #pragma unroll
        for (int i = 0; i < 8; ++i) {
            half4 h = { (_Float16)breg[i][0], (_Float16)breg[i][1],
                        (_Float16)breg[i][2], (_Float16)breg[i][3] };
            *(half4*)(&lds_b[b_row[i] * LDS_STRIDE + b_c4[i]]) = h;
        }

        __syncthreads();

        // ---- issue NEXT K-tile's global loads (overlap with WMMAs below) ----
        if (k0 + BLK_K < K) {
            const int kn = k0 + BLK_K;
            #pragma unroll
            for (int i = 0; i < 4; ++i) areg[i] = *(const f32x4*)(a_ptr[i] + kn);
            #pragma unroll
            for (int i = 0; i < 8; ++i) breg[i] = *(const i32x4*)(b_ptr[i] + kn);
        }

        // ---- fragment loads: contiguous 16 halves (2 x ds_load_b128) --------
        half16 afrag[4], bfrag[4];
        const _Float16* abase = &lds_a[(wm * 64 + r) * LDS_STRIDE + kh * 16];
        const _Float16* bbase = &lds_b[(wn * 64 + r) * LDS_STRIDE + kh * 16];
        #pragma unroll
        for (int mi = 0; mi < 4; ++mi) {
            const _Float16* p = abase + mi * 16 * LDS_STRIDE;
            half8 lo = *(const half8*)(p);
            half8 hi = *(const half8*)(p + 8);
            afrag[mi] = __builtin_shufflevector(lo, hi,
                0,1,2,3,4,5,6,7,8,9,10,11,12,13,14,15);
        }
        #pragma unroll
        for (int ni = 0; ni < 4; ++ni) {
            const _Float16* p = bbase + ni * 16 * LDS_STRIDE;
            half8 lo = *(const half8*)(p);
            half8 hi = *(const half8*)(p + 8);
            bfrag[ni] = __builtin_shufflevector(lo, hi,
                0,1,2,3,4,5,6,7,8,9,10,11,12,13,14,15);
        }

        // ---- 16 WMMAs: 64x64x32 per wave per K step -------------------------
        #pragma unroll
        for (int mi = 0; mi < 4; ++mi)
            #pragma unroll
            for (int ni = 0; ni < 4; ++ni)
                acc[mi][ni] = __builtin_amdgcn_wmma_f32_16x16x32_f16(
                    /*neg_a=*/false, afrag[mi],
                    /*neg_b=*/false, bfrag[ni],
                    /*c_mod=*/(short)0, acc[mi][ni],
                    /*reuse_a=*/false, /*reuse_b=*/false);

        __syncthreads();
    }

    // ---- epilogue: out = acc*scale[n] + bias[n]  (f32) ----------------------
    // C layout (ISA 7.12.2): VGPR v holds M=v (lanes 0-15) / M=v+8 (lanes 16-31),
    // N = lane%16.
    #pragma unroll
    for (int ni = 0; ni < 4; ++ni) {
        int   n  = n0 + wn * 64 + ni * 16 + r;
        float sc = scale[n];
        float bs = bias[n];
        #pragma unroll
        for (int mi = 0; mi < 4; ++mi) {
            int mbase = m0 + wm * 64 + mi * 16 + kh * 8;
            #pragma unroll
            for (int v = 0; v < 8; ++v) {
                out[(size_t)(mbase + v) * N + n] = fmaf(acc[mi][ni][v], sc, bs);
            }
        }
    }
}

extern "C" void kernel_launch(void* const* d_in, const int* in_sizes, int n_in,
                              void* d_out, int out_size, void* d_ws, size_t ws_size,
                              hipStream_t stream) {
    const float* x     = (const float*)d_in[0];   // [4,2048,4096] f32
    const int*   qw    = (const int*)  d_in[1];   // [11008,4096] i32 in [-8,8)
    const float* scale = (const float*)d_in[2];   // [11008,1] f32
    const float* bias  = (const float*)d_in[3];   // [11008] f32
    float*       out   = (float*)d_out;           // [4,2048,11008] f32

    constexpr int M = 4 * 2048;   // 8192
    constexpr int N = 11008;
    constexpr int K = 4096;

    dim3 grid(N / BLK_N, M / BLK_M);   // 43 x 64 workgroups
    bitnet_wmma_f16_kernel<<<grid, 256, 0, stream>>>(x, qw, scale, bias, out,
                                                     M, N, K);
}
